// GAT_30520037605489
// MI455X (gfx1250) — matrix operand order repaired and checked by reading, not compiled, not run on hardware
//
#include <hip/hip_runtime.h>

// ---------------------------------------------------------------------------
// CDNA5 (gfx1250) GAT: bf16 WMMA GEMMs (LDS-staged B, pipelined A and B) +
// L2-resident atomic segment softmax.
// ---------------------------------------------------------------------------

typedef __attribute__((ext_vector_type(16))) __bf16        v16bf;
typedef __attribute__((ext_vector_type(8)))  float         v8f;
typedef __attribute__((ext_vector_type(8)))  unsigned int  u32x8;

#define SLOPE 0.2f

// LDS staging geometry: 64 columns x 256 K-elements per stage.
// Stride padded to 264 elems (528 B): column c starts at bank (c*132)%64 =
// (c*4)%64 -> 16 fragment columns hit 16 distinct 4-bank groups, conflict-free.
#define KSTAGE    256
#define BSTRIDE   264   // KSTAGE + 8

// ---------------- helpers ----------------
__device__ __forceinline__ unsigned short f2bf(float f) {
  unsigned u = __float_as_uint(f);
  return (unsigned short)((u + 0x7FFFu + ((u >> 16) & 1u)) >> 16);  // RNE-ish
}
__device__ __forceinline__ unsigned f2ord(float f) {  // order-preserving map
  unsigned u = __float_as_uint(f);
  return (u & 0x80000000u) ? ~u : (u | 0x80000000u);
}
__device__ __forceinline__ float ord2f(unsigned o) {
  unsigned u = (o & 0x80000000u) ? (o & 0x7FFFFFFFu) : ~o;
  return __uint_as_float(u);
}
__device__ __forceinline__ float lrelu(float s) { return s > 0.f ? s : SLOPE * s; }

__device__ __forceinline__ u32x8 pack2(const uint4 lo, const uint4 hi) {
  u32x8 r;
  r[0] = lo.x; r[1] = lo.y; r[2] = lo.z; r[3] = lo.w;
  r[4] = hi.x; r[5] = hi.y; r[6] = hi.z; r[7] = hi.w;
  return r;
}

// ---------------- conversion kernels ----------------
__global__ void k_f32_to_bf16(const float* __restrict__ in,
                              unsigned short* __restrict__ out, long n) {
  long i = (long)blockIdx.x * blockDim.x + threadIdx.x;
  if (i < n) out[i] = f2bf(in[i]);
}

// W[k*Ncol + c] (f32) -> Wt[c*K + k] (bf16)
__global__ void k_transpose_bf16(const float* __restrict__ W,
                                 unsigned short* __restrict__ Wt,
                                 int K, int Ncol) {
  long i = (long)blockIdx.x * blockDim.x + threadIdx.x;
  if (i >= (long)K * Ncol) return;
  int k = (int)(i / Ncol), c = (int)(i % Ncol);
  Wt[(long)c * K + k] = f2bf(W[i]);
}

// ---------------- WMMA GEMM: C[M,Ncol] = A[M,K](bf16) * Bt[Ncol,K](bf16)T ---
// Block: 8 waves -> 128(M) x 64(N) tile. Wave: 16 x 64 (4 accumulators).
// B strip staged in LDS (KSTAGE-deep halves); A fragment double-buffered from
// global; B fragment double-buffered from LDS (one ds-load pair in flight).
__global__ __launch_bounds__(256) void k_gemm_bf16_wmma(
    const unsigned short* __restrict__ A, const unsigned short* __restrict__ Bt,
    float* __restrict__ C, int M, int K, int Ncol) {
  __shared__ unsigned short bs[64 * BSTRIDE];   // 33,792 B

  const int lane    = threadIdx.x & 31;
  const int wave    = threadIdx.x >> 5;
  const int lane16  = lane & 15;
  const int halfsel = lane >> 4;                 // 0: lanes 0-15, 1: lanes 16-31
  const int row0    = blockIdx.x * 128 + wave * 16;
  const int col0    = blockIdx.y * 64;

  int arow = row0 + lane16; if (arow >= M) arow = M - 1;   // clamp, keep EXEC full
  const unsigned short* Aptr = A + (long)arow * K + (halfsel ? 8 : 0);
  const int kbase2 = halfsel ? 8 : 0;

  // B fragment fetch from LDS: col = lane16 (+ j*16), K = {kb..kb+7} U {16+kb..16+kb+7}.
  auto ldsB = [&](int col16, int kl) -> u32x8 {
    const unsigned short* bp = bs + col16 * BSTRIDE + kl;
    return pack2(*(const uint4*)(bp), *(const uint4*)(bp + 16));
  };

  v8f acc[4] = {};

  // Prime A fragment for k=0.
  // 16-bit A 16x32 layout: per-lane K = {kb..kb+7} U {16+kb..16+kb+7}
  u32x8 a_cur = pack2(*(const uint4*)(Aptr + 0), *(const uint4*)(Aptr + 16));

  for (int khalf = 0; khalf < K; khalf += KSTAGE) {
    __syncthreads();     // previous stage fully consumed
    // Cooperative stage: 64 cols x KSTAGE elems, 16B chunks. 2048 chunks / 256 thr.
    for (int c = threadIdx.x; c < 64 * (KSTAGE / 8); c += 256) {
      const int col  = c >> 5;          // / (KSTAGE/8)
      const int koff = (c & 31) << 3;   // * 8
      *(uint4*)(bs + col * BSTRIDE + koff) =
          *(const uint4*)(Bt + (long)(col0 + col) * K + khalf + koff);
    }
    __syncthreads();

    // Prime first B fragment of this stage (j=0, first k-step).
    u32x8 b_cur = ldsB(lane16, kbase2);

    for (int k0 = khalf; k0 < khalf + KSTAGE; k0 += 32) {
      // Prefetch next K-step's A fragment (register double-buffer).
      const int kn = (k0 + 32 < K) ? (k0 + 32) : k0;
      const u32x8 a_nxt =
          pack2(*(const uint4*)(Aptr + kn), *(const uint4*)(Aptr + kn + 16));

      const v16bf af  = __builtin_bit_cast(v16bf, a_cur);
      const int   kl  = (k0 - khalf) + kbase2;   // local K offset in LDS stage
      // next k-step's frag-0 offset; clamp at stage end (dead value, harmless)
      const int   kln = (k0 + 32 < khalf + KSTAGE) ? (kl + 32) : kbase2;

#pragma unroll
      for (int j = 0; j < 4; ++j) {
        // Prefetch fragment j+1 (or next k-step's fragment 0) one WMMA ahead.
        const u32x8 b_nxt = ldsB(((j + 1) & 3) * 16 + lane16, (j < 3) ? kl : kln);
        acc[j] = __builtin_amdgcn_wmma_f32_16x16x32_bf16(
            false, af, false, __builtin_bit_cast(v16bf, b_cur),
            (short)0, acc[j], false, false);
        b_cur = b_nxt;
      }
      a_cur = a_nxt;
    }
  }

  // C/D layout: lanes 0-15 -> N=lane, M=r; lanes 16-31 -> N=lane-16, M=r+8
  const int rbase = halfsel * 8;
#pragma unroll
  for (int j = 0; j < 4; ++j) {
    const int col = col0 + j * 16 + lane16;
#pragma unroll
    for (int r = 0; r < 8; ++r) {
      const int row = row0 + rbase + r;
      if (row < M) C[(long)row * Ncol + col] = acc[j][r];
    }
  }
}

// ---------------- attention node scores ----------------
// es[n,h] = dot(g[n, h*128 : h*128+128], asrc[h]);  ed likewise.
__global__ void k_node_scores(const float* __restrict__ g,
                              const float* __restrict__ asrc,
                              const float* __restrict__ adst,
                              float* __restrict__ es, float* __restrict__ ed,
                              int N) {
  int i = blockIdx.x * blockDim.x + threadIdx.x;       // i = n*4 + h
  if (i >= N * 4) return;
  int n = i >> 2, h = i & 3;
  const float4* gp = (const float4*)(g + (long)n * 512 + h * 128);
  const float4* ap = (const float4*)(asrc + h * 128);
  const float4* bp = (const float4*)(adst + h * 128);
  float ss = 0.f, sd = 0.f;
#pragma unroll 4
  for (int c = 0; c < 32; ++c) {
    float4 v = gp[c], a = ap[c], b = bp[c];
    ss += v.x * a.x + v.y * a.y + v.z * a.z + v.w * a.w;
    sd += v.x * b.x + v.y * b.y + v.z * b.z + v.w * b.w;
  }
  es[i] = ss; ed[i] = sd;
}

// ---------------- init / copy ----------------
__global__ void k_init_stats(unsigned* __restrict__ mord,
                             float* __restrict__ denom, int n) {
  int i = blockIdx.x * blockDim.x + threadIdx.x;
  if (i < n) { mord[i] = 0u; denom[i] = 0.f; }
}
__global__ void k_copy_f4(float4* __restrict__ dst, const float4* __restrict__ src,
                          long n4) {
  long i = (long)blockIdx.x * blockDim.x + threadIdx.x;
  if (i < n4) dst[i] = src[i];
}

// ---------------- edge passes ----------------
__global__ void k_edge_max(const int* __restrict__ ei,
                           const float* __restrict__ es, const float* __restrict__ ed,
                           unsigned* __restrict__ mord, int E, int Etot) {
  int e = blockIdx.x * blockDim.x + threadIdx.x;
  if (e >= Etot) return;
  int s, d;
  if (e < E) { s = ei[e]; d = ei[E + e]; } else { s = d = e - E; }
#pragma unroll
  for (int h = 0; h < 4; ++h) {
    float sc = lrelu(es[s * 4 + h] + ed[d * 4 + h]);
    atomicMax(&mord[d * 4 + h], f2ord(sc));
  }
}

__global__ void k_edge_expsum(const int* __restrict__ ei,
                              const float* __restrict__ es, const float* __restrict__ ed,
                              const unsigned* __restrict__ mord,
                              float* __restrict__ denom, int E, int Etot) {
  int e = blockIdx.x * blockDim.x + threadIdx.x;
  if (e >= Etot) return;
  int s, d;
  if (e < E) { s = ei[e]; d = ei[E + e]; } else { s = d = e - E; }
#pragma unroll
  for (int h = 0; h < 4; ++h) {
    float sc = lrelu(es[s * 4 + h] + ed[d * 4 + h]);
    float ex = __expf(sc - ord2f(mord[d * 4 + h]));
    atomicAdd(&denom[d * 4 + h], ex);
  }
}

// 128 threads per edge; thread t handles features 4t..4t+3 (single head each).
__global__ __launch_bounds__(256) void k_edge_scatter(
    const int* __restrict__ ei, const float* __restrict__ es,
    const float* __restrict__ ed, const unsigned* __restrict__ mord,
    const float* __restrict__ denom, const float* __restrict__ g,
    float* __restrict__ accum, int E, int Etot) {
  const int t    = threadIdx.x & 127;
  const long e   = (long)blockIdx.x * 2 + (threadIdx.x >> 7);
  if (e >= Etot) return;
  int s, d;
  if (e < E) { s = ei[e]; d = ei[E + e]; } else { s = d = (int)(e - E); }
  const int h = t >> 5;                      // (4t)/128
  float sc    = lrelu(es[s * 4 + h] + ed[d * 4 + h]);
  float alpha = __expf(sc - ord2f(mord[d * 4 + h])) / (denom[d * 4 + h] + 1e-16f);
  const int c = t * 4;
  const float4 gv = *(const float4*)(g + (long)s * 512 + c);
  float* op = accum + (long)d * 512 + c;
  atomicAdd(op + 0, alpha * gv.x);
  atomicAdd(op + 1, alpha * gv.y);
  atomicAdd(op + 2, alpha * gv.z);
  atomicAdd(op + 3, alpha * gv.w);
}

// ---------------- finalize: out = relu(accum + bias) ----------------
__global__ void k_finalize(const float* __restrict__ accum,
                           const float* __restrict__ bias,
                           float* __restrict__ out, long n) {
  long i = ((long)blockIdx.x * blockDim.x + threadIdx.x) * 4;
  if (i >= n) return;
  float4 a = *(const float4*)(accum + i);
  const float4 b = *(const float4*)(bias + (i & 511));
  a.x = fmaxf(a.x + b.x, 0.f);
  a.y = fmaxf(a.y + b.y, 0.f);
  a.z = fmaxf(a.z + b.z, 0.f);
  a.w = fmaxf(a.w + b.w, 0.f);
  *(float4*)(out + i) = a;
}

// ---------------------------------------------------------------------------
extern "C" void kernel_launch(void* const* d_in, const int* in_sizes, int n_in,
                              void* d_out, int out_size, void* d_ws, size_t ws_size,
                              hipStream_t stream) {
  (void)n_in; (void)out_size; (void)ws_size;

  const int N    = in_sizes[0] / 256;
  const int E    = in_sizes[1] / 2;
  const int Etot = E + N;
  const int F    = 512;

  const float* x     = (const float*)d_in[0];
  const int*   ei    = (const int*)d_in[1];
  const float* w0    = (const float*)d_in[2];
  const float* b0    = (const float*)d_in[3];
  const float* as0   = (const float*)d_in[4];
  const float* ad0   = (const float*)d_in[5];
  const float* resw0 = (const float*)d_in[6];
  const float* w1    = (const float*)d_in[7];
  const float* b1    = (const float*)d_in[8];
  const float* as1   = (const float*)d_in[9];
  const float* ad1   = (const float*)d_in[10];
  const float* w2    = (const float*)d_in[11];
  const float* b2    = (const float*)d_in[12];
  const float* as2   = (const float*)d_in[13];
  const float* ad2   = (const float*)d_in[14];

  // workspace carve-up
  size_t off = 0;
  auto carve = [&](size_t bytes) -> void* {
    void* p = (char*)d_ws + off;
    off += (bytes + 255) & ~(size_t)255;
    return p;
  };
  float*          hA    = (float*)carve((size_t)N * F * 4);   // intermediate h
  float*          accum = (float*)carve((size_t)N * F * 4);   // residual + scatter
  float*          g     = (float*)carve((size_t)N * F * 4);   // projected feats
  unsigned short* hbf   = (unsigned short*)carve((size_t)N * F * 2);
  unsigned short* wt    = (unsigned short*)carve((size_t)F * F * 2);
  float*          es    = (float*)carve((size_t)N * 4 * 4);
  float*          ed    = (float*)carve((size_t)N * 4 * 4);
  unsigned*       mord  = (unsigned*)carve((size_t)N * 4 * 4);
  float*          den   = (float*)carve((size_t)N * 4 * 4);

  const long nF   = (long)N * F;
  const int  ebl  = (Etot + 255) / 256;
  const int  nhbl = (N * 4 + 255) / 256;
  const dim3 ggrid((unsigned)((N + 127) / 128), (unsigned)(F / 64));

  auto run_layer = [&](const float* hin, int K, const float* W, const float* bias,
                       const float* as, const float* ad, const float* resW,
                       float* hout) {
    const long nK = (long)N * K;
    // 1) activations -> bf16
    k_f32_to_bf16<<<(unsigned)((nK + 255) / 256), 256, 0, stream>>>(hin, hbf, nK);
    // 2) W -> bf16 transposed [F][K]
    k_transpose_bf16<<<(unsigned)(((long)K * F + 255) / 256), 256, 0, stream>>>(W, wt, K, F);
    // 3) g = h @ W
    k_gemm_bf16_wmma<<<ggrid, 256, 0, stream>>>(hbf, wt, g, N, K, F);
    // 4) accum = residual
    if (resW) {  // layer 0: accum = x @ res_w0 (second WMMA GEMM)
      k_transpose_bf16<<<(unsigned)(((long)K * F + 255) / 256), 256, 0, stream>>>(resW, wt, K, F);
      k_gemm_bf16_wmma<<<ggrid, 256, 0, stream>>>(hbf, wt, accum, N, K, F);
    } else {     // identity residual
      k_copy_f4<<<(unsigned)((nF / 4 + 255) / 256), 256, 0, stream>>>(
          (float4*)accum, (const float4*)hin, nF / 4);
    }
    // 5) node scores
    k_node_scores<<<nhbl, 256, 0, stream>>>(g, as, ad, es, ed, N);
    // 6) segment softmax state init
    k_init_stats<<<nhbl, 256, 0, stream>>>(mord, den, N * 4);
    // 7) segment max / exp-sum / scatter
    k_edge_max<<<ebl, 256, 0, stream>>>(ei, es, ed, mord, E, Etot);
    k_edge_expsum<<<ebl, 256, 0, stream>>>(ei, es, ed, mord, den, E, Etot);
    k_edge_scatter<<<(unsigned)((Etot + 1) / 2), 256, 0, stream>>>(
        ei, es, ed, mord, den, g, accum, E, Etot);
    // 8) hout = relu(accum + bias)
    k_finalize<<<(unsigned)((nF / 4 + 255) / 256), 256, 0, stream>>>(accum, bias, hout, nF);
  };

  float* out = (float*)d_out;
  run_layer(x,  256, w0, b0, as0, ad0, resw0, hA);   // layer 0 (linear residual)
  run_layer(hA, 512, w1, b1, as1, ad1, nullptr, hA); // layer 1 (identity residual)
  run_layer(hA, 512, w2, b2, as2, ad2, nullptr, out);// layer 2 -> d_out
}
